// MultiheadedAttention_62646392979555
// MI455X (gfx1250) — matrix operand order repaired and checked by reading, not compiled, not run on hardware
//
#include <hip/hip_runtime.h>
#include <hip/hip_bf16.h>

// ---------------------------------------------------------------------------
// MI455X (gfx1250) multi-head attention forward.
// - all matmuls: v_wmma_f32_16x16x32_bf16 (wave32), bf16 in / f32 accumulate
// - proj GEMM: double-buffered GLOBAL_LOAD_ASYNC_TO_LDS_B128 staging of W,
//   direct B-fragment reads via DS_LOAD_B128 (packed [n][k] 16x16 tiles)
// - flash attention computes S^T so K feeds the A-side (contiguous loads) and
//   the P^T C-fragment aliases the P A-fragment: zero transposes in the loop
// ---------------------------------------------------------------------------

typedef __attribute__((ext_vector_type(16))) __bf16 v16bf;
typedef __attribute__((ext_vector_type(8)))  __bf16 v8bf;
typedef __attribute__((ext_vector_type(8)))  float  v8f;

constexpr int Bc = 2, Sc = 2048, Dc = 1024, Hc = 16, DKc = 64;
constexpr int Mrows = Bc * Sc;      // 4096

union Frag16 { v16bf v; v8bf h[2]; __bf16 e[16]; };
union FragC  { v8f   v; float e[8]; };
union U8     { v8bf  v; __bf16 e[8]; };

static __device__ __forceinline__ __bf16 f2bf(float f) {
  union { float f; unsigned u; } a; a.f = f;
  unsigned r = (a.u + 0x7FFFu + ((a.u >> 16) & 1u)) >> 16;   // RNE
  union { unsigned short u; __bf16 b; } o; o.u = (unsigned short)r;
  return o.b;
}

static __device__ __forceinline__ v8f wmma_bf16(v16bf a, v16bf b, v8f c) {
  // 8 args: (neg_a, A, neg_b, B, c_mod, C, reuse_a, reuse_b)
  return __builtin_amdgcn_wmma_f32_16x16x32_bf16(false, a, false, b,
                                                 (short)0, c, false, false);
}

// LDS byte offset of a shared-memory pointer (generic addr low 32 bits).
static __device__ __forceinline__ unsigned lds_off(const void* p) {
  return (unsigned)(size_t)p;
}

// ---------------------------------------------------------------------------
// fp32 -> bf16 conversion, 8 elements / thread (b128 in, b128 out)
// ---------------------------------------------------------------------------
__global__ void cvt_f32_bf16(const float* __restrict__ s,
                             __bf16* __restrict__ d, int n) {
  int i = (blockIdx.x * blockDim.x + threadIdx.x) * 8;
  if (i + 8 <= n) {
    float4 a = *(const float4*)(s + i);
    float4 b = *(const float4*)(s + i + 4);
    U8 u;
    u.e[0] = f2bf(a.x); u.e[1] = f2bf(a.y); u.e[2] = f2bf(a.z); u.e[3] = f2bf(a.w);
    u.e[4] = f2bf(b.x); u.e[5] = f2bf(b.y); u.e[6] = f2bf(b.z); u.e[7] = f2bf(b.w);
    *(v8bf*)(d + i) = u.v;
  } else {
    for (; i < n; ++i) d[i] = f2bf(s[i]);
  }
}

// ---------------------------------------------------------------------------
// C[m,n] = X[m,:] . W[n,:] + bias[n]
//   mode 0: scatter bf16 into [B,H,S,DK] projection layout
//   mode 1: write fp32 row-major [Mrows, Dc]
// Block: 256 thr (8 waves). Tile: 128(M) x 64(N). K-step 32.
// W tile staged per K-step as 8 packed 16x16 bf16 tiles (512 B each):
//   tile(f,kt) holds W[nbase+16f .. +15][k0+16kt .. +15], row-major.
// Async copy double-buffers LDS; B-fragments come from plain ds_load_b128.
// ---------------------------------------------------------------------------
__global__ __launch_bounds__(256) void proj_gemm(
    const __bf16* __restrict__ X,      // [Mrows, Dc]
    const __bf16* __restrict__ W,      // [Dc(N), Dc(K)] row-major
    const float*  __restrict__ bias,   // [Dc]
    __bf16* __restrict__ outProj,      // mode 0
    float*  __restrict__ outF32,       // mode 1
    int mode) {
  __shared__ alignas(16) __bf16 ldsW[2][8 * 256];   // 2 x 4 KB packed tiles

  const int tid   = threadIdx.x;
  const int lane  = tid & 31;
  const int wv    = tid >> 5;
  const int mbase = blockIdx.x * 128;
  const int nbase = blockIdx.y * 64;
  const int rlo   = lane & 15;
  const int khalf = (lane >> 4) * 8;
  const int rowA  = mbase + wv * 16 + rlo;

  // Per-thread async-copy coordinates: 16 B (8 halves) per lane per issue.
  const int nl_t = tid >> 2;                 // 0..63  (row within N tile)
  const int q_t  = tid & 3;                  // 0..3   (8-half chunk within 32 K)
  const int f_t  = nl_t >> 4, nsub_t = nl_t & 15;
  const int kt_t = q_t >> 1,  kl8_t  = (q_t & 1) * 8;
  const unsigned dst_t =                     // byte offset within one buffer
      (unsigned)(((f_t * 2 + kt_t) * 256 + nsub_t * 16 + kl8_t) * 2);
  const __bf16* srcW = W + (size_t)(nbase + nl_t) * Dc + q_t * 8;

  FragC acc[4];
#pragma unroll
  for (int f = 0; f < 4; ++f)
#pragma unroll
    for (int r = 0; r < 8; ++r) acc[f].e[r] = 0.0f;

  // Prologue: async-stage K-step 0 into buffer 0.
  {
    unsigned dst = lds_off(&ldsW[0][0]) + dst_t;
    unsigned long long src = (unsigned long long)(size_t)(srcW);
    asm volatile("global_load_async_to_lds_b128 %0, %1, off"
                 :: "v"(dst), "v"(src) : "memory");
  }
  asm volatile("s_wait_asynccnt 0" ::: "memory");
  __syncthreads();

  for (int k0 = 0; k0 < Dc; k0 += 32) {
    const int buf = (k0 >> 5) & 1;

    // Kick async copy of the next W tile into the other buffer.
    if (k0 + 32 < Dc) {
      unsigned dst = lds_off(&ldsW[buf ^ 1][0]) + dst_t;
      unsigned long long src = (unsigned long long)(size_t)(srcW + k0 + 32);
      asm volatile("global_load_async_to_lds_b128 %0, %1, off"
                   :: "v"(dst), "v"(src) : "memory");
    }

    // A fragment: lane = row, contiguous K (ISA 16-bit A 16x32 layout)
    Frag16 a;
    a.h[0] = *(const v8bf*)(X + (size_t)rowA * Dc + k0 + khalf);
    a.h[1] = *(const v8bf*)(X + (size_t)rowA * Dc + k0 + 16 + khalf);
    if (k0 + 32 < Dc)
      __builtin_prefetch(X + (size_t)rowA * Dc + k0 + 32, 0, 3);

    const unsigned bbase = lds_off(&ldsW[buf][0]) + (unsigned)((lane & 15) * 32 + (lane >> 4) * 16);
#pragma unroll
    for (int f = 0; f < 4; ++f) {
      Frag16 bf;
      // Direct read: lane = n row of the [n][k] tile, 8 contiguous k per half-fragment.
      asm volatile("ds_load_b128 %0, %1"
                   : "=v"(bf.h[0]) : "v"(bbase + (unsigned)(f * 2 + 0) * 512u));
      asm volatile("ds_load_b128 %0, %1"
                   : "=v"(bf.h[1]) : "v"(bbase + (unsigned)(f * 2 + 1) * 512u));
      // Enforce DS RAW: thread the fragment through the wait so the WMMA
      // data-depends on s_wait_dscnt.
      asm volatile("s_wait_dscnt 0"
                   : "+v"(bf.h[0]), "+v"(bf.h[1]) :: "memory");
      acc[f].v = wmma_bf16(a.v, bf.v, acc[f].v);
    }

    // My async for buf^1 must land before anyone reads it next iteration.
    asm volatile("s_wait_asynccnt 0" ::: "memory");
    __syncthreads();
  }

#pragma unroll
  for (int f = 0; f < 4; ++f) {
    const int n  = nbase + f * 16 + rlo;
    const float bv = bias[n];
#pragma unroll
    for (int r = 0; r < 8; ++r) {
      const int m = mbase + wv * 16 + (lane >> 4) * 8 + r;  // C layout: row = 8*half+r
      const float val = acc[f].e[r] + bv;
      if (mode == 0) {
        const int b = m >> 11, s = m & (Sc - 1);
        const int h = n >> 6, dk = n & (DKc - 1);
        outProj[(((size_t)b * Hc + h) * Sc + s) * DKc + dk] = f2bf(val);
      } else {
        outF32[(size_t)m * Dc + n] = val;
      }
    }
  }
}

// ---------------------------------------------------------------------------
// Causal flash attention, one 16-row Q tile per wave, 32 keys per iteration.
// Computes S^T = K_tile . Q^T so every global access is contiguous and the
// P^T score fragment aliases the A-fragment of P.V in registers.
// Grid: B*H*(S/128) blocks of 256 threads (8 independent waves).
// ---------------------------------------------------------------------------
__global__ __launch_bounds__(256) void flash_attn(
    const __bf16* __restrict__ Qp,   // [B,H,S,DK] bf16
    const __bf16* __restrict__ Kp,
    const __bf16* __restrict__ Vp,
    __bf16* __restrict__ AttOut) {   // [B,S,D] bf16
  __shared__ alignas(16) __bf16 ldsQt[8][64 * 16];  // per-wave Q^T[k][q], 16 KB

  const int tid  = threadIdx.x;
  const int lane = tid & 31;
  const int wv   = tid >> 5;
  const int tilesPerBH = Sc / 128;                 // 16
  const int bh = blockIdx.x / tilesPerBH;
  const int qt = blockIdx.x % tilesPerBH;
  const int b  = bh / Hc, h = bh % Hc;
  const int qbase = qt * 128 + wv * 16;
  const size_t hb = (size_t)bh * Sc * DKc;

  const int rlo   = lane & 15;
  const int khalf = (lane >> 4) * 8;

  // ---- stage Q^T once: lane loads q-row rlo, dk range [(lane>>4)*32, +32) ----
  {
    const int kb = (lane >> 4) * 32;
    const __bf16* qrow = Qp + hb + (size_t)(qbase + rlo) * DKc + kb;
    U8 u[4];
#pragma unroll
    for (int c = 0; c < 4; ++c) u[c].v = *(const v8bf*)(qrow + c * 8);
#pragma unroll
    for (int c = 0; c < 4; ++c)
#pragma unroll
      for (int i = 0; i < 8; ++i)
        ldsQt[wv][(kb + c * 8 + i) * 16 + rlo] = u[c].e[i];
  }
  asm volatile("s_wait_dscnt 0" ::: "memory");

  // Loop-invariant Q^T B-fragments (2 K-slices of 32 over DK=64)
  Frag16 qb[2];
#pragma unroll
  for (int kk = 0; kk < 2; ++kk) {
    const __bf16* p = &ldsQt[wv][(kk * 32 + khalf) * 16 + rlo];  // B col n = q = lane&15
    for (int i = 0; i < 8; ++i) {
      qb[kk].e[i]     = p[i * 16];
      qb[kk].e[8 + i] = p[(16 + i) * 16];
    }
  }

  float m = -1e30f, l = 0.0f;        // per-lane: stats of query row qbase+rlo
  FragC o[4];
#pragma unroll
  for (int f = 0; f < 4; ++f)
#pragma unroll
    for (int r = 0; r < 8; ++r) o[f].e[r] = 0.0f;

  const float scale = 0.70710678118654752f;  // 1/sqrt(batch) per reference
  const int qglob = qbase + rlo;
  const int qend  = qbase + 15;

  for (int j0 = 0; j0 <= qend; j0 += 32) {
    // S^T tiles: st0 = keys j0..j0+15, st1 = keys j0+16..j0+31 (M = key, N = q)
    FragC st0, st1;
#pragma unroll
    for (int r = 0; r < 8; ++r) { st0.e[r] = 0.0f; st1.e[r] = 0.0f; }
#pragma unroll
    for (int kk = 0; kk < 2; ++kk) {
      Frag16 ka0, ka1;  // A-side: K rows, contiguous dk
      const __bf16* kr0 = Kp + hb + (size_t)(j0 + rlo) * DKc + kk * 32;
      const __bf16* kr1 = Kp + hb + (size_t)(j0 + 16 + rlo) * DKc + kk * 32;
      ka0.h[0] = *(const v8bf*)(kr0 + khalf);
      ka0.h[1] = *(const v8bf*)(kr0 + 16 + khalf);
      ka1.h[0] = *(const v8bf*)(kr1 + khalf);
      ka1.h[1] = *(const v8bf*)(kr1 + 16 + khalf);
      st0.v = wmma_bf16(ka0.v, qb[kk].v, st0.v);
      st1.v = wmma_bf16(ka1.v, qb[kk].v, st1.v);
    }

    // Scale + causal mask; row stats are per-lane (q = lane&15)
    float t = -1e30f;
#pragma unroll
    for (int r = 0; r < 8; ++r) {
      const int key0 = j0 + khalf + r;         // st0 element key index
      const int key1 = j0 + 16 + khalf + r;    // st1 element key index
      float v0 = st0.e[r] * scale;
      float v1 = st1.e[r] * scale;
      if (key0 > qglob) v0 = -1e30f;
      if (key1 > qglob) v1 = -1e30f;
      st0.e[r] = v0; st1.e[r] = v1;
      t = fmaxf(t, fmaxf(v0, v1));
    }
    t = fmaxf(t, __shfl_xor(t, 16, 32));       // combine the two lane halves

    const float mnew = fmaxf(m, t);
    const float corr = __expf(m - mnew);
    float rs = 0.0f;
    Frag16 pa;                                  // P A-fragment, built in regs
#pragma unroll
    for (int r = 0; r < 8; ++r) {
      const float p0 = __expf(st0.e[r] - mnew);
      const float p1 = __expf(st1.e[r] - mnew);
      rs += p0 + p1;
      pa.e[r]     = f2bf(p0);                   // key = khalf + r
      pa.e[8 + r] = f2bf(p1);                   // key = 16 + khalf + r
    }
    rs += __shfl_xor(rs, 16, 32);
    l = l * corr + rs;
    m = mnew;

    // Rescale O: O rows are r-indexed (q = 8*(lane>>4)+r), stats lane-indexed
#pragma unroll
    for (int r = 0; r < 8; ++r) {
      const float c = __shfl(corr, (lane >> 4) * 8 + r, 32);
#pragma unroll
      for (int f = 0; f < 4; ++f) o[f].e[r] *= c;
    }

    // O += P . V ; V B-fragments gathered (lane = dk column, elements = keys)
    const __bf16* vrow = Vp + hb + (size_t)(j0 + khalf) * DKc + rlo;
#pragma unroll
    for (int f = 0; f < 4; ++f) {
      Frag16 vb;
      for (int i = 0; i < 8; ++i) {
        vb.e[i]     = vrow[(size_t)i * DKc + f * 16];
        vb.e[8 + i] = vrow[(size_t)(16 + i) * DKc + f * 16];
      }
      o[f].v = wmma_bf16(pa.v, vb.v, o[f].v);
    }
  }

  // Normalize and store to [B,S,D] bf16
  const float invl = 1.0f / l;
#pragma unroll
  for (int r = 0; r < 8; ++r) {
    const float inv = __shfl(invl, (lane >> 4) * 8 + r, 32);
    const int mq = qbase + (lane >> 4) * 8 + r;
    const size_t orow = ((size_t)b * Sc + mq) * Dc + (size_t)h * DKc;
#pragma unroll
    for (int f = 0; f < 4; ++f)
      AttOut[orow + f * 16 + rlo] = f2bf(o[f].e[r] * inv);
  }
}

// ---------------------------------------------------------------------------
// Launch
// ---------------------------------------------------------------------------
extern "C" void kernel_launch(void* const* d_in, const int* in_sizes, int n_in,
                              void* d_out, int out_size, void* d_ws, size_t ws_size,
                              hipStream_t stream) {
  const float* q  = (const float*)d_in[0];
  const float* k  = (const float*)d_in[1];
  const float* v  = (const float*)d_in[2];
  // d_in[3]: causal mask (tril) — structure applied analytically in flash_attn
  const float* Wq = (const float*)d_in[4];
  const float* Wk = (const float*)d_in[5];
  const float* Wv = (const float*)d_in[6];
  const float* Wo = (const float*)d_in[7];
  const float* bq = (const float*)d_in[8];
  const float* bk = (const float*)d_in[9];
  const float* bv = (const float*)d_in[10];
  const float* bo = (const float*)d_in[11];

  char* ws = (char*)d_ws;
  const size_t SZX = (size_t)Mrows * Dc * sizeof(__bf16);  // 8 MB
  const size_t SZW = (size_t)Dc * Dc * sizeof(__bf16);     // 2 MB
  __bf16* xq  = (__bf16*)(ws);
  __bf16* xk  = (__bf16*)(ws + SZX);
  __bf16* xv  = (__bf16*)(ws + 2 * SZX);
  __bf16* wqb = (__bf16*)(ws + 3 * SZX);
  __bf16* wkb = (__bf16*)(ws + 3 * SZX + SZW);
  __bf16* wvb = (__bf16*)(ws + 3 * SZX + 2 * SZW);
  __bf16* wob = (__bf16*)(ws + 3 * SZX + 3 * SZW);
  __bf16* qp  = (__bf16*)(ws + 3 * SZX + 4 * SZW);
  __bf16* kp  = (__bf16*)(ws + 4 * SZX + 4 * SZW);
  __bf16* vp  = (__bf16*)(ws + 5 * SZX + 4 * SZW);
  __bf16* ao  = (__bf16*)(ws + 6 * SZX + 4 * SZW);

  const int nX = Mrows * Dc;   // 4,194,304
  const int nW = Dc * Dc;      // 1,048,576
  const int cvtBX = (nX / 8 + 255) / 256;
  const int cvtBW = (nW / 8 + 255) / 256;
  cvt_f32_bf16<<<cvtBX, 256, 0, stream>>>(q, xq, nX);
  cvt_f32_bf16<<<cvtBX, 256, 0, stream>>>(k, xk, nX);
  cvt_f32_bf16<<<cvtBX, 256, 0, stream>>>(v, xv, nX);
  cvt_f32_bf16<<<cvtBW, 256, 0, stream>>>(Wq, wqb, nW);
  cvt_f32_bf16<<<cvtBW, 256, 0, stream>>>(Wk, wkb, nW);
  cvt_f32_bf16<<<cvtBW, 256, 0, stream>>>(Wv, wvb, nW);
  cvt_f32_bf16<<<cvtBW, 256, 0, stream>>>(Wo, wob, nW);

  dim3 g(Mrows / 128, Dc / 64);   // 32 x 16
  proj_gemm<<<g, 256, 0, stream>>>(xq, wqb, bq, qp, nullptr, 0);
  proj_gemm<<<g, 256, 0, stream>>>(xk, wkb, bk, kp, nullptr, 0);
  proj_gemm<<<g, 256, 0, stream>>>(xv, wvb, bv, vp, nullptr, 0);

  flash_attn<<<Bc * Hc * (Sc / 128), 256, 0, stream>>>(qp, kp, vp, ao);

  proj_gemm<<<g, 256, 0, stream>>>(ao, wob, bo, nullptr, (float*)d_out, 1);
}
